// Block_16844861735766
// MI455X (gfx1250) — matrix operand order repaired and checked by reading, not compiled, not run on hardware
//
#include <hip/hip_runtime.h>
#include <hip/hip_bf16.h>

// Transformer block (B=4, T=2048, D=64, H=8) for MI455X (gfx1250, wave32).
// ~37 GFLOP vs tens of MB of traffic -> WMMA-math bound. f16 WMMA with f32
// accumulation; flash attention so T x T scores never reach HBM. All matrix
// operands live in "fragment-ready" swizzled layouts so every fragment load
// is b128-vectorized. Weights are converted f32->f16+swizzled ONCE by a prep
// kernel; per-block staging is then a pure async global->LDS byte copy
// (GLOBAL_LOAD_ASYNC_TO_LDS_B128, ASYNCcnt) with no VGPR round trip.

#define NE 64      // n_embd / head size
#define NH 8       // heads
#define NB 4       // batch
#define NT 2048    // seq len
#define HDIM 512   // NH*NE
#define DFF 256    // 4*NE
#define MTOT (NB*NT)

typedef __attribute__((ext_vector_type(16))) _Float16 v16h;
typedef __attribute__((ext_vector_type(8)))  _Float16 v8h;
typedef __attribute__((ext_vector_type(8)))  float    v8f;

__device__ inline v8f wmma_f16(v16h a, v16h b, v8f c) {
  return __builtin_amdgcn_wmma_f32_16x16x32_f16(false, a, false, b, (short)0, c,
                                                false, false);
}

__device__ inline v16h cat16(v8h lo, v8h hi) {
  return __builtin_shufflevector(lo, hi, 0, 1, 2, 3, 4, 5, 6, 7, 8, 9, 10, 11,
                                 12, 13, 14, 15);
}

// A-frag (16x32 MxK) from row-major f16 (ld halfs): two 16-byte loads.
__device__ inline v16h load_a16(const _Float16* p, int ld) {
  const int lane = threadIdx.x & 31;
  const _Float16* q = p + (lane & 15) * ld + ((lane >> 4) << 4);
  return cat16(*(const v8h*)q, *(const v8h*)(q + 8));
}

// Fragment from a swizzled tile: lane L's 16 halfs contiguous at tile[L*16].
__device__ inline v16h load_frag_swz(const _Float16* tile) {
  const int lane = threadIdx.x & 31;
  const _Float16* q = tile + lane * 16;
  return cat16(*(const v8h*)q, *(const v8h*)(q + 8));
}

// Swizzled-B element address within a 32x16 (KxN) tile of 512 halfs:
//   lane = ((k>>4)&1)*16 + n ; elem = k & 15
__device__ inline int bswz(int k, int n) {
  return ((((k >> 4) & 1) << 4) + n) * 16 + (k & 15);
}

// Async 16-byte global->LDS copy (per-lane addresses), tracked by ASYNCcnt.
__device__ inline void async_copy16(unsigned lds_byte_addr, const void* gaddr) {
  asm volatile("global_load_async_to_lds_b128 %0, %1, off"
               :: "v"(lds_byte_addr), "v"(gaddr) : "memory");
}
__device__ inline unsigned lds_addr_of(const void* p) {
  return (unsigned)(unsigned long long)p;  // low 32 bits = LDS offset
}
__device__ inline void wait_async_and_barrier() {
  asm volatile("s_wait_asynccnt 0" ::: "memory");
  __syncthreads();
}

// 16-lane-group reductions (xor masks 1..8 stay within each wave32 half).
__device__ inline float red_max16(float x) {
#pragma unroll
  for (int m = 1; m <= 8; m <<= 1) x = fmaxf(x, __shfl_xor(x, m, 32));
  return x;
}
__device__ inline float red_sum16(float x) {
#pragma unroll
  for (int m = 1; m <= 8; m <<= 1) x += __shfl_xor(x, m, 32);
  return x;
}

// ---------------------------------------------------------------------------
// K0: one-shot weight conversion f32 -> f16 B-swizzled fragment layout.
//   qkvw: 24 matrices of [64][64]: idx (m*NH+h), tiles [k/32][n/16]
//   wpw : [512][64]  -> 16x4 tiles ; w1w: [64][256] -> 2x16 ; w2w: [256][64]
__global__ __launch_bounds__(256) void prep_weights_kernel(
    const float* __restrict__ Wq, const float* __restrict__ Wk,
    const float* __restrict__ Wv, const float* __restrict__ Wp,
    const float* __restrict__ W1, const float* __restrict__ W2,
    _Float16* __restrict__ qkvw, _Float16* __restrict__ wpw,
    _Float16* __restrict__ w1w, _Float16* __restrict__ w2w) {
  const int gid = blockIdx.x * 256 + threadIdx.x;
  const int nthr = gridDim.x * 256;
  for (int idx = gid; idx < 3 * NH * NE * NE; idx += nthr) {
    const int mh = idx >> 12, e = idx & 4095;
    const int k = e >> 6, n = e & 63;
    const float* src = (mh < NH) ? Wq : ((mh < 2 * NH) ? Wk : Wv);
    const float v = src[(size_t)(mh & 7) * 4096 + e];
    qkvw[(size_t)mh * 4096 + ((k >> 5) * 4 + (n >> 4)) * 512 +
         bswz(k, n & 15)] = (_Float16)v;
  }
  for (int idx = gid; idx < HDIM * NE; idx += nthr) {
    const int k = idx >> 6, n = idx & 63;
    wpw[((k >> 5) * 4 + (n >> 4)) * 512 + bswz(k, n & 15)] = (_Float16)Wp[idx];
  }
  for (int idx = gid; idx < NE * DFF; idx += nthr) {
    const int k = idx >> 8, n = idx & 255;
    w1w[((k >> 5) * 16 + (n >> 4)) * 512 + bswz(k, n & 15)] = (_Float16)W1[idx];
  }
  for (int idx = gid; idx < DFF * NE; idx += nthr) {
    const int k = idx >> 6, n = idx & 63;
    w2w[((k >> 5) * 4 + (n >> 4)) * 512 + bswz(k, n & 15)] = (_Float16)W2[idx];
  }
}

// ---------------------------------------------------------------------------
// K1: per-head QKV projection.  grid (NT/128, NB, NH), 256 threads (8 waves).
// Q row-major (A-frag source); K score-B-swizzled; V PV-B-swizzled (whose
// store is contiguous: one v8h store per tile).
__global__ __launch_bounds__(256) void qkv_kernel(
    const float* __restrict__ x, const _Float16* __restrict__ qkvw,
    const float* __restrict__ bq, const float* __restrict__ bk,
    const float* __restrict__ bv,
    _Float16* __restrict__ qf, _Float16* __restrict__ kf,
    _Float16* __restrict__ vf) {
  __shared__ __align__(16) _Float16 xs[128 * NE];     // 16 KB, row-major
  __shared__ __align__(16) _Float16 wl[3 * NE * NE];  // 24 KB, pre-swizzled
  const int tile = blockIdx.x, bb = blockIdx.y, h = blockIdx.z;
  const int tid = threadIdx.x;
  const int row0 = tile * 128;

  // async byte-copy of the three pre-swizzled weight matrices into LDS
  const _Float16* wq_s = qkvw + (size_t)h * 4096;
  const _Float16* wk_s = qkvw + (size_t)(NH + h) * 4096;
  const _Float16* wv_s = qkvw + (size_t)(2 * NH + h) * 4096;
#pragma unroll
  for (int i = 0; i < 2; ++i) {  // 512 chunks x 8 halfs per matrix
    const int c = (tid + i * 256) * 8;
    async_copy16(lds_addr_of(&wl[c]), wq_s + c);
    async_copy16(lds_addr_of(&wl[4096 + c]), wk_s + c);
    async_copy16(lds_addr_of(&wl[8192 + c]), wv_s + c);
  }
  // x tile needs f32->f16 conversion: classic staging
  const float4* xb = (const float4*)(x + ((size_t)bb * NT + row0) * NE);
#pragma unroll
  for (int i = 0; i < 8; ++i) {  // 2048 float4 / 256
    const int idx = tid + i * 256;
    const float4 f = xb[idx];
    _Float16* d = &xs[idx * 4];
    d[0] = (_Float16)f.x; d[1] = (_Float16)f.y;
    d[2] = (_Float16)f.z; d[3] = (_Float16)f.w;
  }
  wait_async_and_barrier();

  const int w = tid >> 5, lane = tid & 31;
  const int col = lane & 15, mb = (lane >> 4) * 8;
  const int jrow0 = row0 + w * 16;  // this wave's 16 rows (global in T)
  v16h a0 = load_a16(&xs[(w * 16) * NE], NE);
  v16h a1 = load_a16(&xs[(w * 16) * NE + 32], NE);
  const size_t hbase = (size_t)(h * NB + bb) * NT * NE;  // halfs

  // ---- Q: row-major [T][64] ----
#pragma unroll
  for (int n = 0; n < 4; ++n) {
    v8f c = {};
    c = wmma_f16(a0, load_frag_swz(&wl[n * 512]), c);
    c = wmma_f16(a1, load_frag_swz(&wl[(4 + n) * 512]), c);
    const float bvv = bq[h * NE + n * 16 + col];
    _Float16* op = qf + hbase + (size_t)jrow0 * NE;
#pragma unroll
    for (int i = 0; i < 8; ++i)
      op[(mb + i) * NE + n * 16 + col] = (_Float16)(c[i] + bvv);
  }
  // ---- K: swizzled for q@k^T B-frags: tile = (key>>4)*2 + (dim>>5) ----
#pragma unroll
  for (int n = 0; n < 4; ++n) {
    v8f c = {};
    c = wmma_f16(a0, load_frag_swz(&wl[4096 + n * 512]), c);
    c = wmma_f16(a1, load_frag_swz(&wl[4096 + (4 + n) * 512]), c);
    const int d = n * 16 + col;
    const float bvv = bk[h * NE + d];
    const size_t tb = hbase + (size_t)((jrow0 >> 4) * 2 + (d >> 5)) * 512 +
                      ((((d >> 4) & 1) << 4) << 4) + (d & 15);
#pragma unroll
    for (int i = 0; i < 8; ++i)
      kf[tb + (size_t)(mb + i) * 16] = (_Float16)(c[i] + bvv);
  }
  // ---- V: swizzled for P@V B-frags: per-lane rows contiguous -> v8h store --
#pragma unroll
  for (int n = 0; n < 4; ++n) {
    v8f c = {};
    c = wmma_f16(a0, load_frag_swz(&wl[8192 + n * 512]), c);
    c = wmma_f16(a1, load_frag_swz(&wl[8192 + (4 + n) * 512]), c);
    const int d = n * 16 + col;
    const float bvv = bv[h * NE + d];
    v8h vo;
#pragma unroll
    for (int i = 0; i < 8; ++i) vo[i] = (_Float16)(c[i] + bvv);
    const size_t off = hbase + (size_t)((jrow0 >> 5) * 4 + (d >> 4)) * 512 +
                       (size_t)((((jrow0 >> 4) & 1) << 4) + col) * 16 + mb;
    *(v8h*)(vf + off) = vo;
  }
}

// ---------------------------------------------------------------------------
// K2: causal flash attention (unscaled scores).  grid (NT/128, NB, NH).
// Waves have different causal trip counts -> no block barriers; P round-trips
// a per-wave LDS slab in A-frag-swizzled form (in-order DS + s_wait_dscnt).
__global__ __launch_bounds__(256) void attn_kernel(
    const _Float16* __restrict__ qf, const _Float16* __restrict__ kf,
    const _Float16* __restrict__ vf, _Float16* __restrict__ cat) {
  __shared__ __align__(16) _Float16 plds[8][512];  // one 16x32 P tile per wave
  const int tile = blockIdx.x, bb = blockIdx.y, h = blockIdx.z;
  const int tid = threadIdx.x, w = tid >> 5, lane = tid & 31;
  const int col = lane & 15, mb = (lane >> 4) * 8;
  const int r0 = tile * 128 + w * 16;
  const size_t hb = (size_t)(h * NB + bb) * NT;
  const _Float16* qb = qf + (hb + r0) * NE;
  const _Float16* kswz = kf + hb * NE;
  const _Float16* vswz = vf + hb * NE;

  v16h aq0 = load_a16(qb, NE);
  v16h aq1 = load_a16(qb + 32, NE);
  v8f o0 = {}, o1 = {}, o2 = {}, o3 = {};
  float mi[8], li[8];
#pragma unroll
  for (int i = 0; i < 8; ++i) { mi[i] = -3.0e38f; li[i] = 0.0f; }

  // P A-frag swizzle: P(r,k) -> (r + (k>>4)*16)*16 + (k&15)
  const int pw0 = mb * 16 + col;

  const int kend = r0 + 16;  // causal: keys [0, r0+16)
  for (int k0 = 0; k0 < kend; k0 += 32) {
    const int jt = k0 >> 4;
    v8f s0 = {}, s1 = {};
    s0 = wmma_f16(aq0, load_frag_swz(kswz + (size_t)(jt * 2 + 0) * 512), s0);
    s0 = wmma_f16(aq1, load_frag_swz(kswz + (size_t)(jt * 2 + 1) * 512), s0);
    s1 = wmma_f16(aq0, load_frag_swz(kswz + (size_t)(jt * 2 + 2) * 512), s1);
    s1 = wmma_f16(aq1, load_frag_swz(kswz + (size_t)(jt * 2 + 3) * 512), s1);
    __builtin_prefetch(kswz + (size_t)(jt * 2 + 4) * 512, 0, 3);
    __builtin_prefetch(vswz + (size_t)((k0 >> 5) + 1) * 4 * 512, 0, 3);

#pragma unroll
    for (int i = 0; i < 8; ++i) {
      const int qi = r0 + mb + i;
      s0[i] = (k0 + col > qi) ? -3.0e38f : s0[i];
      s1[i] = (k0 + 16 + col > qi) ? -3.0e38f : s1[i];
    }

    float sc[8];
#pragma unroll
    for (int i = 0; i < 8; ++i) {
      const float rowmax = red_max16(fmaxf(s0[i], s1[i]));
      const float mn = fmaxf(mi[i], rowmax);
      sc[i] = __expf(mi[i] - mn);
      mi[i] = mn;
    }
#pragma unroll
    for (int i = 0; i < 8; ++i) {
      const float p0 = __expf(s0[i] - mi[i]);
      const float p1 = __expf(s1[i] - mi[i]);
      li[i] = li[i] * sc[i] + red_sum16(p0 + p1);
      o0[i] *= sc[i]; o1[i] *= sc[i]; o2[i] *= sc[i]; o3[i] *= sc[i];
      plds[w][pw0 + i * 16] = (_Float16)p0;      // k = col
      plds[w][pw0 + i * 16 + 256] = (_Float16)p1;  // k = 16+col
    }
    // wave-local LDS turnaround: DS is in-order per wave; wait + compiler fence
    asm volatile("s_wait_dscnt 0" ::: "memory");
    v16h ap = load_frag_swz(&plds[w][0]);  // 2 x ds_load_b128
    const _Float16* vt = vswz + (size_t)(k0 >> 5) * 4 * 512;
    o0 = wmma_f16(ap, load_frag_swz(vt), o0);
    o1 = wmma_f16(ap, load_frag_swz(vt + 512), o1);
    o2 = wmma_f16(ap, load_frag_swz(vt + 1024), o2);
    o3 = wmma_f16(ap, load_frag_swz(vt + 1536), o3);
  }

  // normalize, write concat layout [B, T, H*NE] f16 (row-major: proj A source)
  const size_t cbase = ((size_t)bb * NT + r0) * HDIM + (size_t)h * NE;
#pragma unroll
  for (int i = 0; i < 8; ++i) {
    const float inv = 1.0f / li[i];
    _Float16* cp = cat + cbase + (size_t)(mb + i) * HDIM;
    cp[col] = (_Float16)(o0[i] * inv);
    cp[16 + col] = (_Float16)(o1[i] * inv);
    cp[32 + col] = (_Float16)(o2[i] * inv);
    cp[48 + col] = (_Float16)(o3[i] * inv);
  }
}

// ---------------------------------------------------------------------------
// K3/K5: out = layernorm(resid + A@Wsw + bias) (64 cols).  grid (MTOT/128).
// Wsw is pre-swizzled f16 -> staging is a pure async global->LDS copy.
__global__ __launch_bounds__(256) void gemm_ln_kernel(
    const _Float16* __restrict__ A, const _Float16* __restrict__ Wsw,
    const float* __restrict__ bias, const float* __restrict__ resid,
    const float* __restrict__ gamma, const float* __restrict__ beta,
    float* __restrict__ out32, _Float16* __restrict__ out16, int K) {
  __shared__ __align__(16) _Float16 Wl[HDIM * NE];  // 64 KB max (K<=512)
  const int tid = threadIdx.x;
  for (int c = tid * 8; c < K * NE; c += 256 * 8)
    async_copy16(lds_addr_of(&Wl[c]), Wsw + c);
  wait_async_and_barrier();

  const int w = tid >> 5, lane = tid & 31;
  const int col = lane & 15, mb = (lane >> 4) * 8;
  const int r0 = blockIdx.x * 128 + w * 16;

  v8f c0 = {}, c1 = {}, c2 = {}, c3 = {};
  for (int k0 = 0; k0 < K; k0 += 32) {
    v16h a = load_a16(A + (size_t)r0 * K + k0, K);
    const _Float16* tb = &Wl[(size_t)(k0 >> 5) * 4 * 512];
    c0 = wmma_f16(a, load_frag_swz(tb), c0);
    c1 = wmma_f16(a, load_frag_swz(tb + 512), c1);
    c2 = wmma_f16(a, load_frag_swz(tb + 1024), c2);
    c3 = wmma_f16(a, load_frag_swz(tb + 1536), c3);
  }

  const float bb0 = bias[col], bb1 = bias[16 + col];
  const float bb2 = bias[32 + col], bb3 = bias[48 + col];
  const float g0 = gamma[col], g1v = gamma[16 + col];
  const float g2v = gamma[32 + col], g3 = gamma[48 + col];
  const float e0 = beta[col], e1 = beta[16 + col];
  const float e2 = beta[32 + col], e3 = beta[48 + col];
#pragma unroll
  for (int i = 0; i < 8; ++i) {
    const size_t r = (size_t)(r0 + mb + i) * NE;
    float x0 = c0[i] + bb0 + resid[r + col];
    float x1 = c1[i] + bb1 + resid[r + 16 + col];
    float x2 = c2[i] + bb2 + resid[r + 32 + col];
    float x3 = c3[i] + bb3 + resid[r + 48 + col];
    const float mu = red_sum16(x0 + x1 + x2 + x3) * (1.0f / 64.0f);
    x0 -= mu; x1 -= mu; x2 -= mu; x3 -= mu;
    const float var =
        red_sum16(x0 * x0 + x1 * x1 + x2 * x2 + x3 * x3) * (1.0f / 64.0f);
    const float rs = rsqrtf(var + 1e-5f);
    const float y0 = x0 * rs * g0 + e0, y1 = x1 * rs * g1v + e1;
    const float y2 = x2 * rs * g2v + e2, y3 = x3 * rs * g3 + e3;
    out32[r + col] = y0; out32[r + 16 + col] = y1;
    out32[r + 32 + col] = y2; out32[r + 48 + col] = y3;
    if (out16) {
      out16[r + col] = (_Float16)y0; out16[r + 16 + col] = (_Float16)y1;
      out16[r + 32 + col] = (_Float16)y2; out16[r + 48 + col] = (_Float16)y3;
    }
  }
}

// ---------------------------------------------------------------------------
// K4: hidden = relu(h1 @ W1 + b1), [MTOT, 256] f16.  grid (MTOT/128).
__global__ __launch_bounds__(256) void ffn1_kernel(
    const _Float16* __restrict__ A, const _Float16* __restrict__ W1s,
    const float* __restrict__ b1, _Float16* __restrict__ hid) {
  __shared__ __align__(16) _Float16 Wl[NE * DFF];  // 32 KB pre-swizzled
  const int tid = threadIdx.x;
#pragma unroll
  for (int i = 0; i < 8; ++i) {  // 2048 chunks x 8 halfs
    const int c = (tid + i * 256) * 8;
    async_copy16(lds_addr_of(&Wl[c]), W1s + c);
  }
  wait_async_and_barrier();

  const int w = tid >> 5, lane = tid & 31;
  const int col = lane & 15, mb = (lane >> 4) * 8;
  const int r0 = blockIdx.x * 128 + w * 16;
  v16h a0 = load_a16(A + (size_t)r0 * NE, NE);
  v16h a1 = load_a16(A + (size_t)r0 * NE + 32, NE);

#pragma unroll
  for (int nc = 0; nc < 4; ++nc) {  // 64-column chunks of the 256 outputs
    const int n0 = nc * 64;
    v8f c0 = {}, c1 = {}, c2 = {}, c3 = {};
    c0 = wmma_f16(a0, load_frag_swz(&Wl[(nc * 4 + 0) * 512]), c0);
    c0 = wmma_f16(a1, load_frag_swz(&Wl[(16 + nc * 4 + 0) * 512]), c0);
    c1 = wmma_f16(a0, load_frag_swz(&Wl[(nc * 4 + 1) * 512]), c1);
    c1 = wmma_f16(a1, load_frag_swz(&Wl[(16 + nc * 4 + 1) * 512]), c1);
    c2 = wmma_f16(a0, load_frag_swz(&Wl[(nc * 4 + 2) * 512]), c2);
    c2 = wmma_f16(a1, load_frag_swz(&Wl[(16 + nc * 4 + 2) * 512]), c2);
    c3 = wmma_f16(a0, load_frag_swz(&Wl[(nc * 4 + 3) * 512]), c3);
    c3 = wmma_f16(a1, load_frag_swz(&Wl[(16 + nc * 4 + 3) * 512]), c3);
    const float bb0 = b1[n0 + col], bb1 = b1[n0 + 16 + col];
    const float bb2 = b1[n0 + 32 + col], bb3 = b1[n0 + 48 + col];
#pragma unroll
    for (int i = 0; i < 8; ++i) {
      _Float16* hp = hid + (size_t)(r0 + mb + i) * DFF + n0;
      hp[col] = (_Float16)fmaxf(c0[i] + bb0, 0.0f);
      hp[16 + col] = (_Float16)fmaxf(c1[i] + bb1, 0.0f);
      hp[32 + col] = (_Float16)fmaxf(c2[i] + bb2, 0.0f);
      hp[48 + col] = (_Float16)fmaxf(c3[i] + bb3, 0.0f);
    }
  }
}

// ---------------------------------------------------------------------------
extern "C" void kernel_launch(void* const* d_in, const int* in_sizes, int n_in,
                              void* d_out, int out_size, void* d_ws,
                              size_t ws_size, hipStream_t stream) {
  const float* x  = (const float*)d_in[0];
  const float* Wq = (const float*)d_in[1];
  const float* bq = (const float*)d_in[2];
  const float* Wk = (const float*)d_in[3];
  const float* bk = (const float*)d_in[4];
  const float* Wv = (const float*)d_in[5];
  const float* bv = (const float*)d_in[6];
  const float* Wp = (const float*)d_in[7];
  const float* bp = (const float*)d_in[8];
  const float* W1 = (const float*)d_in[9];
  const float* b1 = (const float*)d_in[10];
  const float* W2 = (const float*)d_in[11];
  const float* b2 = (const float*)d_in[12];
  const float* g1 = (const float*)d_in[13];
  const float* be1 = (const float*)d_in[14];
  const float* g2 = (const float*)d_in[15];
  const float* be2 = (const float*)d_in[16];

  char* p = (char*)d_ws;
  _Float16* qf = (_Float16*)p;   p += (size_t)NH * NB * NT * NE * 2;  // 8 MB
  _Float16* kf = (_Float16*)p;   p += (size_t)NH * NB * NT * NE * 2;  // 8 MB
  _Float16* vf = (_Float16*)p;   p += (size_t)NH * NB * NT * NE * 2;  // 8 MB
  _Float16* cat = (_Float16*)p;  p += (size_t)MTOT * HDIM * 2;        // 8 MB
  float* h1 = (float*)p;         p += (size_t)MTOT * NE * 4;          // 2 MB
  _Float16* h1h = (_Float16*)p;  p += (size_t)MTOT * NE * 2;          // 1 MB
  _Float16* hid = (_Float16*)p;  p += (size_t)MTOT * DFF * 2;         // 4 MB
  _Float16* qkvw = (_Float16*)p; p += (size_t)3 * NH * NE * NE * 2;   // 192 KB
  _Float16* wpw = (_Float16*)p;  p += (size_t)HDIM * NE * 2;          // 64 KB
  _Float16* w1w = (_Float16*)p;  p += (size_t)NE * DFF * 2;           // 32 KB
  _Float16* w2w = (_Float16*)p;  p += (size_t)DFF * NE * 2;           // 32 KB

  const dim3 blk(256);
  prep_weights_kernel<<<dim3(96), blk, 0, stream>>>(Wq, Wk, Wv, Wp, W1, W2,
                                                    qkvw, wpw, w1w, w2w);
  qkv_kernel<<<dim3(NT / 128, NB, NH), blk, 0, stream>>>(x, qkvw, bq, bk, bv,
                                                         qf, kf, vf);
  attn_kernel<<<dim3(NT / 128, NB, NH), blk, 0, stream>>>(qf, kf, vf, cat);
  gemm_ln_kernel<<<dim3(MTOT / 128), blk, 0, stream>>>(
      cat, wpw, bp, /*resid=*/x, g1, be1, h1, h1h, HDIM);
  ffn1_kernel<<<dim3(MTOT / 128), blk, 0, stream>>>(h1h, w1w, b1, hid);
  gemm_ln_kernel<<<dim3(MTOT / 128), blk, 0, stream>>>(
      hid, w2w, b2, /*resid=*/h1, g2, be2, (float*)d_out, nullptr, DFF);
}
